// MultiHeadSelfAttentionEdges_14130442404001
// MI455X (gfx1250) — matrix operand order repaired and checked by reading, not compiled, not run on hardware
//
#include <hip/hip_runtime.h>
#include <hip/hip_bf16.h>

typedef __bf16 bf16_t;
typedef __bf16 v16bf __attribute__((ext_vector_type(16)));
typedef float  v8f   __attribute__((ext_vector_type(8)));

#define DIMC    768
#define NHEADS  12
#define NNODES  256
#define NBATCH  2
#define HDIM    64
#define QKSCALE 0.125f
#define NEGINF  (-32768.0f)

// ---------------------------------------------------------------- WMMA core
__device__ __forceinline__ v8f wmma_bf16(v16bf a, v16bf b, v8f c) {
  return __builtin_amdgcn_wmma_f32_16x16x32_bf16(false, a, false, b, (short)0, c,
                                                 false, false);
}

// A operand (16x32 bf16): lane holds row (lane&15); K set = {kb..kb+7, kb+16..kb+23},
// kb = k0 + 8*(lane>=16). Source row is K-contiguous f32.
__device__ __forceinline__ v16bf fragA_f32(const float* row, int k0, int lane) {
  const float* p = row + k0 + ((lane >> 4) & 1) * 8;
  v16bf f;
#pragma unroll
  for (int i = 0; i < 8; ++i) f[i] = (bf16_t)p[i];
#pragma unroll
  for (int i = 0; i < 8; ++i) f[8 + i] = (bf16_t)p[16 + i];
  return f;
}
// B operand (32x16 bf16): lane holds column (lane&15); K set = {kb..kb+7, kb+16..kb+23},
// kb = k0 + 8*(lane>=16). Source "row" = K-contiguous data of that column.
__device__ __forceinline__ v16bf fragB_f32(const float* row, int k0, int lane) {
  const float* p = row + k0 + ((lane >> 4) & 1) * 8;
  v16bf f;
#pragma unroll
  for (int i = 0; i < 8; ++i) f[i] = (bf16_t)p[i];
#pragma unroll
  for (int i = 0; i < 8; ++i) f[8 + i] = (bf16_t)p[16 + i];
  return f;
}
__device__ __forceinline__ v16bf fragB_bf16(const bf16_t* row, int k0, int lane) {
  const bf16_t* p = row + k0 + ((lane >> 4) & 1) * 8;
  v16bf f;
#pragma unroll
  for (int i = 0; i < 8; ++i) f[i] = p[i];
#pragma unroll
  for (int i = 0; i < 8; ++i) f[8 + i] = p[16 + i];
  return f;
}

// ------------------------------------------------------- 1) edge PE tables
// P0 = pe @ W1a^T + e_b1 ; P1 = pe @ W1b^T   (each padded to 272 x 768)
__global__ void k_tables(const float* __restrict__ pe, const float* __restrict__ e_w1,
                         const float* __restrict__ e_b1, float* __restrict__ P0,
                         float* __restrict__ P1) {
  int wave = (blockIdx.x * blockDim.x + threadIdx.x) >> 5;
  int lane = threadIdx.x & 31;
  int table = wave / (17 * 48);
  int rem   = wave % (17 * 48);
  int mt = rem / 48, nt = rem % 48;

  int arow = mt * 16 + (lane & 15);
  if (arow > 256) arow = 256;                       // clamp (rows 257..271 are pad)
  int dcol = nt * 16 + (lane & 15);
  const float* A    = pe + (long)arow * DIMC;
  const float* Brow = e_w1 + (long)dcol * (2 * DIMC) + table * DIMC;

  v8f c = {};
  for (int k0 = 0; k0 < DIMC; k0 += 32)
    c = wmma_bf16(fragA_f32(A, k0, lane), fragB_f32(Brow, k0, lane), c);

  float* Pout = (table == 0) ? P0 : P1;
  int row0 = mt * 16 + 8 * ((lane >> 4) & 1);
  int col  = nt * 16 + (lane & 15);
  float bias = (table == 0) ? e_b1[col] : 0.0f;
#pragma unroll
  for (int r = 0; r < 8; ++r)
    Pout[(long)(row0 + r) * DIMC + col] = c[r] + bias;
}

// ------------------------------------------------------------- 2) QKV GEMM
__global__ void k_qkv(const float* __restrict__ x, const float* __restrict__ qkv_w,
                      float* __restrict__ qkv) {
  int wave = (blockIdx.x * blockDim.x + threadIdx.x) >> 5;
  int lane = threadIdx.x & 31;
  int mt = wave / 144, nt = wave % 144;             // M=512, N=2304
  const float* A    = x + (long)(mt * 16 + (lane & 15)) * DIMC;
  const float* Brow = qkv_w + (long)(nt * 16 + (lane & 15)) * DIMC;
  v8f c = {};
  for (int k0 = 0; k0 < DIMC; k0 += 32)
    c = wmma_bf16(fragA_f32(A, k0, lane), fragB_f32(Brow, k0, lane), c);
  int row0 = mt * 16 + 8 * ((lane >> 4) & 1);
  int col  = nt * 16 + (lane & 15);
#pragma unroll
  for (int r = 0; r < 8; ++r)
    qkv[(long)(row0 + r) * 2304 + col] = c[r];
}

// -------------------------------------------- 3) edge scores e2[b,n,m,h]
// e2 = relu(P0[i0] + P1[i1]) @ e_w2^T + e_b2  (16 pairs x 16 heads tile)
__global__ void k_edge(const int* __restrict__ edges, const float* __restrict__ P0,
                       const float* __restrict__ P1, const float* __restrict__ e_w2,
                       const float* __restrict__ e_b2, float* __restrict__ e2out) {
  __shared__ float  ew2f[NHEADS * DIMC];            // 36 KB raw f32 (async DMA dest)
  __shared__ bf16_t ew2s[16 * DIMC];                // 24 KB zero-padded bf16 operand

  // Bulk-stage e_w2 via gfx1250 async global->LDS (ASYNCcnt path, bypasses VGPRs)
  {
    unsigned lds0 = (unsigned)(size_t)(void*)ew2f;  // LDS aperture: addr[31:0]
    for (int i = threadIdx.x; i < (NHEADS * DIMC) / 4; i += blockDim.x) {
      unsigned     lds = lds0 + (unsigned)i * 16u;
      const float* gp  = e_w2 + (long)i * 4;
      asm volatile("global_load_async_to_lds_b128 %0, %1, off"
                   :: "v"(lds), "v"(gp) : "memory");
    }
    asm volatile("s_wait_asynccnt 0" ::: "memory");
  }
  __syncthreads();
  for (int i = threadIdx.x; i < 16 * DIMC; i += blockDim.x) {
    int r = i / DIMC;
    ew2s[i] = (r < NHEADS) ? (bf16_t)ew2f[i] : (bf16_t)0.0f;
  }
  __syncthreads();

  int wave = (blockIdx.x * blockDim.x + threadIdx.x) >> 5;
  int lane = threadIdx.x & 31;
  long pair0 = (long)wave * 16;
  long p = pair0 + (lane & 15);
  int i0 = edges[p * 2 + 0];
  int i1 = edges[p * 2 + 1];
  i0 = (i0 < 0) ? 0 : ((i0 > 256) ? 256 : i0);
  i1 = (i1 < 0) ? 0 : ((i1 > 256) ? 256 : i1);
  const float*  r0   = P0 + (long)i0 * DIMC;
  const float*  r1   = P1 + (long)i1 * DIMC;
  __builtin_prefetch(r0, 0, 3);                     // global_prefetch_b8 (L2-resident
  __builtin_prefetch(r1, 0, 3);                     //  gather rows -> WGP level)
  const bf16_t* brow = ew2s + (lane & 15) * DIMC;
  int kb = ((lane >> 4) & 1) * 8;

  v8f c = {};
  for (int k0 = 0; k0 < DIMC; k0 += 32) {
    int base = k0 + kb;
    v16bf a;
#pragma unroll
    for (int i = 0; i < 8; ++i) {
      float v = r0[base + i] + r1[base + i];
      a[i] = (bf16_t)(v > 0.0f ? v : 0.0f);
    }
#pragma unroll
    for (int i = 0; i < 8; ++i) {
      float v = r0[base + 16 + i] + r1[base + 16 + i];
      a[8 + i] = (bf16_t)(v > 0.0f ? v : 0.0f);
    }
    c = wmma_bf16(a, fragB_bf16(brow, k0, lane), c);
  }
  int h = lane & 15;
  long prow0 = pair0 + 8 * ((lane >> 4) & 1);
  if (h < NHEADS) {
    float bias = e_b2[h];
#pragma unroll
    for (int r = 0; r < 8; ++r)
      e2out[(prow0 + r) * NHEADS + h] = c[r] + bias;
  }
}

// ------------------------------------------- 4) fused attention per (b,h)
__global__ void k_attn(const float* __restrict__ qkv, const float* __restrict__ e2,
                       const unsigned char* __restrict__ mask,
                       float* __restrict__ aout) {
  __shared__ bf16_t Vt[HDIM][NNODES];               // V^T, 32 KB
  __shared__ float  Pb[4][16][NNODES];              // logits/probs, 64 KB
  __shared__ float  rinvS[4][16];

  int b = blockIdx.x / NHEADS;
  int h = blockIdx.x % NHEADS;
  int wave = threadIdx.x >> 5;
  int lane = threadIdx.x & 31;
  int half = (lane >> 4) & 1;
  int l15  = lane & 15;

  for (int i = threadIdx.x; i < HDIM * NNODES; i += blockDim.x) {
    int m = i >> 6, hd = i & 63;                    // coalesced over hd
    Vt[hd][m] = (bf16_t)qkv[(long)(b * NNODES + m) * 2304 + 1536 + h * 64 + hd];
  }
  __syncthreads();

  for (int rt = wave; rt < 16; rt += 4) {           // every wave: 4 row-tiles
    int n0 = rt * 16;
    const float* Qrow = qkv + (long)(b * NNODES + n0 + l15) * 2304 + h * 64;

    float rmax[8];
#pragma unroll
    for (int r = 0; r < 8; ++r) rmax[r] = -3.0e38f;

    // ---- S = Q K^T * scale + e2, masked; stash raw logits in LDS
    for (int mt = 0; mt < 16; ++mt) {
      const float* Krow =
          qkv + (long)(b * NNODES + mt * 16 + l15) * 2304 + 768 + h * 64;
      v8f c = {};
      for (int k0 = 0; k0 < HDIM; k0 += 32)
        c = wmma_bf16(fragA_f32(Qrow, k0, lane), fragB_f32(Krow, k0, lane), c);
      int col = mt * 16 + l15;
#pragma unroll
      for (int r = 0; r < 8; ++r) {
        int n = n0 + r + 8 * half;
        long pidx = (long)(b * NNODES + n) * NNODES + col;
        float s = c[r] * QKSCALE + e2[pidx * NHEADS + h];
        s = mask[pidx] ? s : NEGINF;
        rmax[r] = fmaxf(rmax[r], s);
        Pb[wave][r + 8 * half][col] = s;
      }
    }
    // ---- row max across the 16 lanes of each half (wave32 shuffles)
#pragma unroll
    for (int r = 0; r < 8; ++r)
#pragma unroll
      for (int off = 1; off < 16; off <<= 1)
        rmax[r] = fmaxf(rmax[r], __shfl_xor(rmax[r], off, 32));

    // ---- exp + row sum (lane-private LDS slots)
    float rsum[8];
#pragma unroll
    for (int r = 0; r < 8; ++r) rsum[r] = 0.0f;
    for (int mt = 0; mt < 16; ++mt) {
      int col = mt * 16 + l15;
#pragma unroll
      for (int r = 0; r < 8; ++r) {
        float pv = __expf(Pb[wave][r + 8 * half][col] - rmax[r]);
        Pb[wave][r + 8 * half][col] = pv;
        rsum[r] += pv;
      }
    }
#pragma unroll
    for (int r = 0; r < 8; ++r)
#pragma unroll
      for (int off = 1; off < 16; off <<= 1)
        rsum[r] += __shfl_xor(rsum[r], off, 32);
    if (l15 == 0) {
#pragma unroll
      for (int r = 0; r < 8; ++r) rinvS[wave][r + 8 * half] = 1.0f / rsum[r];
    }
    asm volatile("s_wait_dscnt 0" ::: "memory");    // CDNA5 split LDS counter
    float rinv = rinvS[wave][l15];

    // ---- O = softmax(S) @ V  (1/rowsum folded into A-fragment)
    const float* Prow = &Pb[wave][l15][0];
    v8f oc[4] = {};
    for (int k0 = 0; k0 < NNODES; k0 += 32) {
      int base = k0 + 8 * half;
      v16bf a;
#pragma unroll
      for (int i = 0; i < 8; ++i) a[i] = (bf16_t)(Prow[base + i] * rinv);
#pragma unroll
      for (int i = 0; i < 8; ++i) a[8 + i] = (bf16_t)(Prow[base + 16 + i] * rinv);
#pragma unroll
      for (int nt = 0; nt < 4; ++nt)
        oc[nt] = wmma_bf16(a, fragB_bf16(&Vt[nt * 16 + l15][0], k0, lane), oc[nt]);
    }
#pragma unroll
    for (int nt = 0; nt < 4; ++nt) {
      int hd = nt * 16 + l15;
#pragma unroll
      for (int r = 0; r < 8; ++r) {
        int n = n0 + r + 8 * half;
        aout[(long)(b * NNODES + n) * DIMC + h * 64 + hd] = oc[nt][r];
      }
    }
    asm volatile("s_wait_dscnt 0" ::: "memory");
  }
}

// ------------------------------------------------------- 5) out projection
__global__ void k_oproj(const float* __restrict__ ain, const float* __restrict__ out_w,
                        const float* __restrict__ out_b, float* __restrict__ out) {
  int wave = (blockIdx.x * blockDim.x + threadIdx.x) >> 5;
  int lane = threadIdx.x & 31;
  int mt = wave / 48, nt = wave % 48;               // M=512, N=768
  const float* A    = ain + (long)(mt * 16 + (lane & 15)) * DIMC;
  const float* Brow = out_w + (long)(nt * 16 + (lane & 15)) * DIMC;
  v8f c = {};
  for (int k0 = 0; k0 < DIMC; k0 += 32)
    c = wmma_bf16(fragA_f32(A, k0, lane), fragB_f32(Brow, k0, lane), c);
  int row0 = mt * 16 + 8 * ((lane >> 4) & 1);
  int col  = nt * 16 + (lane & 15);
  float bias = out_b[col];
#pragma unroll
  for (int r = 0; r < 8; ++r)
    out[(long)(row0 + r) * DIMC + col] = c[r] + bias;
}

// --------------------------------------------------------------- launcher
extern "C" void kernel_launch(void* const* d_in, const int* in_sizes, int n_in,
                              void* d_out, int out_size, void* d_ws, size_t ws_size,
                              hipStream_t stream) {
  const float* x     = (const float*)d_in[0];
  const int*   edges = (const int*)d_in[1];
  const unsigned char* mask = (const unsigned char*)d_in[2];
  const float* pe    = (const float*)d_in[3];
  const float* qkv_w = (const float*)d_in[4];
  const float* e_w1  = (const float*)d_in[5];
  const float* e_b1  = (const float*)d_in[6];
  const float* e_w2  = (const float*)d_in[7];
  const float* e_b2  = (const float*)d_in[8];
  const float* out_w = (const float*)d_in[9];
  const float* out_b = (const float*)d_in[10];

  float* out = (float*)d_out;                   // (2,256,768)
  float* e2  = out + 512 * DIMC;                // (2,256,256,12)

  float* ws   = (float*)d_ws;
  float* P0   = ws;                             // 272*768
  float* P1   = P0 + 272 * DIMC;                // 272*768
  float* qkv  = P1 + 272 * DIMC;                // 512*2304
  float* aout = qkv + 512 * 2304;               // 512*768

  k_tables<<<204, 256, 0, stream>>>(pe, e_w1, e_b1, P0, P1);   // 2*17*48 tiles
  k_qkv   <<<576, 256, 0, stream>>>(x, qkv_w, qkv);            // 32*144 tiles
  k_edge  <<<1024, 256, 0, stream>>>(edges, P0, P1, e_w2, e_b2, e2); // 8192 tiles
  k_attn  <<<24, 128, 0, stream>>>(qkv, e2, mask, aout);       // one (b,h)/block
  k_oproj <<<192, 256, 0, stream>>>(aout, out_w, out_b, out);  // 32*48 tiles
}
